// Block_19576460935680
// MI455X (gfx1250) — compile-verified, matrix operand order checked by reading, not run on hardware
//
#include <hip/hip_runtime.h>
#include <hip/hip_bf16.h>
#include <math.h>

typedef __bf16 bf16_t;
typedef __bf16 v8bf  __attribute__((ext_vector_type(8)));
typedef __bf16 v16bf __attribute__((ext_vector_type(16)));
typedef float  v8f   __attribute__((ext_vector_type(8)));
typedef int    v4i   __attribute__((ext_vector_type(4)));

// Pointer-to-int4 in global (AS1) / LDS (AS3) address spaces.
typedef __attribute__((address_space(1))) v4i* gptr_v4i;
typedef __attribute__((address_space(3))) v4i* lptr_v4i;

// ---------------------------------------------------------------------------
// Async global->LDS staging (CDNA5 GLOBAL_LOAD_ASYNC_TO_LDS_B128, ASYNCcnt).
// Guarded: falls back to a synchronous copy if the builtin is absent.
// ---------------------------------------------------------------------------
#if defined(__has_builtin)
#if __has_builtin(__builtin_amdgcn_global_load_async_to_lds_b128)
#define HAS_ASYNC_LDS 1
#endif
#endif
#ifndef HAS_ASYNC_LDS
#define HAS_ASYNC_LDS 0
#endif

__device__ __forceinline__ void stage16(const bf16_t* g, bf16_t* l) {
#if HAS_ASYNC_LDS
  __builtin_amdgcn_global_load_async_to_lds_b128((gptr_v4i)g, (lptr_v4i)l, 0, 0);
#else
  *(v8bf*)l = *(const v8bf*)g;
#endif
}

#if HAS_ASYNC_LDS
#if defined(__has_builtin) && __has_builtin(__builtin_amdgcn_s_wait_asynccnt)
#define ASYNC_WAIT(n) __builtin_amdgcn_s_wait_asynccnt(n)
#else
#define ASYNC_WAIT(n) asm volatile("s_wait_asynccnt %0" ::"i"(n) : "memory")
#endif
#else
#define ASYNC_WAIT(n) ((void)0)
#endif

// ---------------------------------------------------------------------------
// WMMA helpers (V_WMMA_F32_16X16X32_BF16, wave32)
// ---------------------------------------------------------------------------
__device__ __forceinline__ v8f wmma_bf16(v16bf a, v16bf b, v8f c) {
  return __builtin_amdgcn_wmma_f32_16x16x32_bf16(false, a, false, b, (short)0, c,
                                                 false, false);
}

__device__ __forceinline__ v8f zero8() {
  v8f z;
#pragma unroll
  for (int i = 0; i < 8; ++i) z[i] = 0.0f;
  return z;
}

// A (16x32) / B (32x16) bf16 fragment: lane half h holds K {8h..8h+7} U
// {16+8h..16+8h+7} of its row/column (row/col = lane%16). 16B-aligned base.
__device__ __forceinline__ v16bf frag_ab(const bf16_t* rowbase, int half) {
  v8bf lo = *(const v8bf*)(rowbase + 8 * half);
  v8bf hi = *(const v8bf*)(rowbase + 16 + 8 * half);
  v16bf r;
#pragma unroll
  for (int i = 0; i < 8; ++i) { r[i] = lo[i]; r[i + 8] = hi[i]; }
  return r;
}

// ---------------------------------------------------------------------------
// Weight conversion fp32 [K][N] -> bf16 transposed [N][K] (contiguous K rows,
// so GEMM B-tiles are pure row copies -> async-LDS stageable).
// ---------------------------------------------------------------------------
__global__ __launch_bounds__(256) void cvt_transpose_bf16(
    const float* __restrict__ in, bf16_t* __restrict__ out, int K, int N) {
  int i = blockIdx.x * 256 + threadIdx.x;
  if (i < K * N) {
    int n = i / K, k = i - n * K;
    out[i] = (bf16_t)in[(size_t)k * N + n];
  }
}

// ---------------------------------------------------------------------------
// LayerNorm (row of 768) -> bf16
// ---------------------------------------------------------------------------
__global__ __launch_bounds__(256) void ln_bf16_kernel(const float* __restrict__ x,
                                                      const float* __restrict__ w,
                                                      const float* __restrict__ b,
                                                      bf16_t* __restrict__ out) {
  const int C = 768;
  const int row = blockIdx.x;
  const int tid = threadIdx.x;
  const float* xr = x + (size_t)row * C;
  float v0 = xr[tid], v1 = xr[tid + 256], v2 = xr[tid + 512];
  float s = v0 + v1 + v2;
  float s2 = v0 * v0 + v1 * v1 + v2 * v2;
#pragma unroll
  for (int o = 16; o > 0; o >>= 1) {
    s += __shfl_xor(s, o, 32);
    s2 += __shfl_xor(s2, o, 32);
  }
  __shared__ float red[2][8];
  if ((tid & 31) == 0) { red[0][tid >> 5] = s; red[1][tid >> 5] = s2; }
  __syncthreads();
  float ts = 0.f, ts2 = 0.f;
#pragma unroll
  for (int i = 0; i < 8; ++i) { ts += red[0][i]; ts2 += red[1][i]; }
  float mean = ts * (1.0f / 768.0f);
  float var = ts2 * (1.0f / 768.0f) - mean * mean;
  float rstd = rsqrtf(var + 1e-5f);
  bf16_t* orow = out + (size_t)row * C;
  orow[tid]       = (bf16_t)((v0 - mean) * rstd * w[tid] + b[tid]);
  orow[tid + 256] = (bf16_t)((v1 - mean) * rstd * w[tid + 256] + b[tid + 256]);
  orow[tid + 512] = (bf16_t)((v2 - mean) * rstd * w[tid + 512] + b[tid + 512]);
}

// ---------------------------------------------------------------------------
// bf16 GEMM, B pre-transposed ([N][K]). Block tile 128x128, BK=32, 8 waves
// (4 along M x 2 along N), 32x64 per wave = 8 WMMA / K-step. Double-buffered
// LDS fed by async global->LDS copies.
// ---------------------------------------------------------------------------
#define GBM 128
#define GBN 128
#define GBK 32
#define GLDA 40  // elems; 80B row stride (16B multiple)
#define GLDB 40

enum { MODE_QKV = 0, MODE_PROJ = 1, MODE_FC1 = 2, MODE_FC2 = 3 };

__global__ __launch_bounds__(256) void gemm_bf16_kernel(
    const bf16_t* __restrict__ A, const bf16_t* __restrict__ Bt,  // Bt: [N][K]
    int M, int N, int K, int mode,
    const float* __restrict__ bias, const float* __restrict__ resid,
    float* __restrict__ outf, bf16_t* __restrict__ outbf,
    bf16_t* __restrict__ qout, bf16_t* __restrict__ kout, bf16_t* __restrict__ vout) {
  __shared__ bf16_t As[2][GBM * GLDA];
  __shared__ bf16_t Bs[2][GBN * GLDB];
  const int tid = threadIdx.x;
  const int lane = tid & 31, wid = tid >> 5;
  const int half = lane >> 4, l16 = lane & 15;
  const int wm = (wid & 3) * 32;
  const int wn = (wid >> 2) * 64;
  const int m0 = blockIdx.x * GBM;
  const int n0 = blockIdx.y * GBN;

  const int sr = tid >> 1, sc = (tid & 1) * 16;  // 128 rows x 32 cols staging

  v8f acc[2][4];
#pragma unroll
  for (int i = 0; i < 2; ++i)
#pragma unroll
    for (int j = 0; j < 4; ++j) acc[i][j] = zero8();

  auto stage = [&](int buf, int k0) {
    const bf16_t* ga = A + (size_t)(m0 + sr) * K + k0 + sc;
    stage16(ga, &As[buf][sr * GLDA + sc]);
    stage16(ga + 8, &As[buf][sr * GLDA + sc + 8]);
    const bf16_t* gb = Bt + (size_t)(n0 + sr) * K + k0 + sc;
    stage16(gb, &Bs[buf][sr * GLDB + sc]);
    stage16(gb + 8, &Bs[buf][sr * GLDB + sc + 8]);
  };

  stage(0, 0);
  const int nk = K / GBK;
  for (int it = 0; it < nk; ++it) {
    const int buf = it & 1;
    if (it + 1 < nk) {
      stage(buf ^ 1, (it + 1) * GBK);  // overlap next tile with this compute
      ASYNC_WAIT(4);                   // tile `it` complete (in-order), 4 in flight
    } else {
      ASYNC_WAIT(0);
    }
    __syncthreads();

    v16bf af0 = frag_ab(&As[buf][(wm + l16) * GLDA], half);
    v16bf af1 = frag_ab(&As[buf][(wm + 16 + l16) * GLDA], half);
#pragma unroll
    for (int j = 0; j < 4; ++j) {
      v16bf bfj = frag_ab(&Bs[buf][(wn + j * 16 + l16) * GLDB], half);
      acc[0][j] = wmma_bf16(af0, bfj, acc[0][j]);
      acc[1][j] = wmma_bf16(af1, bfj, acc[1][j]);
    }
    __syncthreads();
  }

#pragma unroll
  for (int i = 0; i < 2; ++i)
#pragma unroll
    for (int j = 0; j < 4; ++j)
#pragma unroll
      for (int r = 0; r < 8; ++r) {
        int row = m0 + wm + i * 16 + r + 8 * half;  // C layout
        int col = n0 + wn + j * 16 + l16;
        float v = acc[i][j][r];
        if (mode == MODE_QKV) {
          int which = col / 768;
          int hh = (col % 768) >> 6;
          int dd = col & 63;
          int bb = row >> 10, tok = row & 1023;
          bf16_t bv = (bf16_t)v;
          if (which == 0)
            qout[(((size_t)bb * 12 + hh) * 1024 + tok) * 64 + dd] = bv;
          else if (which == 1)
            kout[(((size_t)bb * 12 + hh) * 1024 + tok) * 64 + dd] = bv;
          else  // V stored transposed [B,H,D,N] for async attention staging
            vout[(((size_t)bb * 12 + hh) * 64 + dd) * 1024 + tok] = bv;
        } else if (mode == MODE_PROJ) {
          size_t idx = (size_t)row * N + col;
          outf[idx] = v + resid[idx];
        } else if (mode == MODE_FC1) {
          float t = v + bias[col];
          float g = 0.5f * t * (1.0f + erff(t * 0.70710678118654752f));
          outbf[(size_t)row * N + col] = (bf16_t)g;
        } else {  // MODE_FC2
          size_t idx = (size_t)row * N + col;
          outf[idx] = v + bias[col] + resid[idx];
        }
      }
}

// ---------------------------------------------------------------------------
// Flash attention: block per (b,h, 128-query tile), 8 waves x 16 rows.
// S = (Q K^T) * 8, online softmax, O = P V. Key tile 64; K and V^T tiles are
// async-staged (V is pre-transposed to [B,H,D,N] by the QKV epilogue).
// ---------------------------------------------------------------------------
#define AQT 128
#define AKT 64
#define HD 64
#define ALDK 72
#define ALDV 72
#define ALDP 72

__global__ __launch_bounds__(256) void attn_kernel(const bf16_t* __restrict__ Q,
                                                   const bf16_t* __restrict__ Km,
                                                   const bf16_t* __restrict__ Vt_g,
                                                   bf16_t* __restrict__ O) {
  const int N = 1024;
  __shared__ bf16_t Ks[AKT * ALDK];     // [key][d]
  __shared__ bf16_t Vt[HD * ALDV];      // [d][key]
  __shared__ bf16_t Ps[8 * 16 * ALDP];  // per-wave P pane
  const int tid = threadIdx.x;
  const int lane = tid & 31, wid = tid >> 5;
  const int half = lane >> 4, l16 = lane & 15;

  const int bh = blockIdx.x >> 3;  // b*12 + h
  const int qt = blockIdx.x & 7;
  const bf16_t* Qb = Q + (size_t)bh * N * HD;
  const bf16_t* Kb = Km + (size_t)bh * N * HD;
  const bf16_t* Vb = Vt_g + (size_t)bh * HD * N;  // [d][tok]
  const int q0 = qt * AQT + wid * 16;

  v16bf qf0, qf1;
  {
    const bf16_t* qrow = Qb + (size_t)(q0 + l16) * HD;
    qf0 = frag_ab(qrow, half);
    qf1 = frag_ab(qrow + 32, half);
  }

  v8f oacc[4];
#pragma unroll
  for (int dt = 0; dt < 4; ++dt) oacc[dt] = zero8();
  float rmax[8], rsum[8];
#pragma unroll
  for (int r = 0; r < 8; ++r) { rmax[r] = -3.0e38f; rsum[r] = 0.0f; }

  bf16_t* pw = Ps + wid * 16 * ALDP;
  const int sr = tid >> 2, sc = (tid & 3) * 16;  // 64 rows x 64 cols staging

  for (int kt0 = 0; kt0 < N; kt0 += AKT) {
    // async stage: K tile rows (key-major) and V^T tile rows (d-major)
    {
      const bf16_t* gk = Kb + (size_t)(kt0 + sr) * HD + sc;
      stage16(gk, Ks + sr * ALDK + sc);
      stage16(gk + 8, Ks + sr * ALDK + sc + 8);
      const bf16_t* gv = Vb + (size_t)sr * N + kt0 + sc;
      stage16(gv, Vt + sr * ALDV + sc);
      stage16(gv + 8, Vt + sr * ALDV + sc + 8);
    }
    ASYNC_WAIT(0);
    __syncthreads();

    // S = Q K^T, 4 column tiles of 16 keys
    v8f s[4];
#pragma unroll
    for (int ct = 0; ct < 4; ++ct) {
      s[ct] = zero8();
      const bf16_t* krow = Ks + (ct * 16 + l16) * ALDK;
      s[ct] = wmma_bf16(qf0, frag_ab(krow, half), s[ct]);
      s[ct] = wmma_bf16(qf1, frag_ab(krow + 32, half), s[ct]);
    }

    float cmax[8];
#pragma unroll
    for (int r = 0; r < 8; ++r) cmax[r] = rmax[r];
#pragma unroll
    for (int ct = 0; ct < 4; ++ct)
#pragma unroll
      for (int r = 0; r < 8; ++r) {
        float v = s[ct][r] * 8.0f;  // reference multiplies by sqrt(D)=8
        s[ct][r] = v;
        cmax[r] = fmaxf(cmax[r], v);
      }
#pragma unroll
    for (int o = 1; o < 16; o <<= 1)
#pragma unroll
      for (int r = 0; r < 8; ++r)
        cmax[r] = fmaxf(cmax[r], __shfl_xor(cmax[r], o, 32));

    float psum[8];
#pragma unroll
    for (int r = 0; r < 8; ++r) {
      float alpha = __expf(rmax[r] - cmax[r]);
      rmax[r] = cmax[r];
      rsum[r] *= alpha;
#pragma unroll
      for (int dt = 0; dt < 4; ++dt) oacc[dt][r] *= alpha;
      psum[r] = 0.0f;
    }
#pragma unroll
    for (int ct = 0; ct < 4; ++ct)
#pragma unroll
      for (int r = 0; r < 8; ++r) {
        float p = __expf(s[ct][r] - rmax[r]);
        psum[r] += p;
        pw[(r + 8 * half) * ALDP + ct * 16 + l16] = (bf16_t)p;
      }
#pragma unroll
    for (int o = 1; o < 16; o <<= 1)
#pragma unroll
      for (int r = 0; r < 8; ++r) psum[r] += __shfl_xor(psum[r], o, 32);
#pragma unroll
    for (int r = 0; r < 8; ++r) rsum[r] += psum[r];

    // same-wave LDS RAW: P-pane stores feed the A-frag loads below
    asm volatile("s_wait_dscnt 0" ::: "memory");

    v16bf pf0 = frag_ab(pw + l16 * ALDP, half);
    v16bf pf1 = frag_ab(pw + l16 * ALDP + 32, half);
#pragma unroll
    for (int dt = 0; dt < 4; ++dt) {
      const bf16_t* vrow = Vt + (dt * 16 + l16) * ALDV;
      oacc[dt] = wmma_bf16(pf0, frag_ab(vrow, half), oacc[dt]);
      oacc[dt] = wmma_bf16(pf1, frag_ab(vrow + 32, half), oacc[dt]);
    }
    __syncthreads();
  }

  const int b = bh / 12, h = bh % 12;
#pragma unroll
  for (int dt = 0; dt < 4; ++dt)
#pragma unroll
    for (int r = 0; r < 8; ++r) {
      int tok = q0 + r + 8 * half;
      float v = oacc[dt][r] / rsum[r];
      O[((size_t)(b * 1024 + tok)) * 768 + h * 64 + dt * 16 + l16] = (bf16_t)v;
    }
}

// ---------------------------------------------------------------------------
// Host orchestration
// ---------------------------------------------------------------------------
static inline void* ws_take(char*& p, size_t bytes) {
  void* r = (void*)p;
  p += (bytes + 255) & ~(size_t)255;
  return r;
}

extern "C" void kernel_launch(void* const* d_in, const int* in_sizes, int n_in,
                              void* d_out, int out_size, void* d_ws, size_t ws_size,
                              hipStream_t stream) {
  (void)in_sizes; (void)n_in; (void)out_size; (void)ws_size;
  const float* x    = (const float*)d_in[0];
  const float* ln1w = (const float*)d_in[1];
  const float* ln1b = (const float*)d_in[2];
  const float* ln2w = (const float*)d_in[3];
  const float* ln2b = (const float*)d_in[4];
  const float* wqkv = (const float*)d_in[5];
  const float* wprj = (const float*)d_in[6];
  const float* wfc1 = (const float*)d_in[7];
  const float* bfc1 = (const float*)d_in[8];
  const float* wfc2 = (const float*)d_in[9];
  const float* bfc2 = (const float*)d_in[10];
  float* out = (float*)d_out;

  const int M = 8192;
  const size_t TOKC = (size_t)M * 768;

  char* p = (char*)d_ws;
  bf16_t* h1 = (bf16_t*)ws_take(p, TOKC * 2);  // LN1 out; reused as attn out
  bf16_t* qb = (bf16_t*)ws_take(p, TOKC * 2);  // q; reused as LN2 out
  bf16_t* kb = (bf16_t*)ws_take(p, TOKC * 2);  // k; reused as GELU out
  bf16_t* vb = (bf16_t*)ws_take(p, TOKC * 2);  // v (transposed [B,H,D,N])
  float*  x1 = (float*)ws_take(p, TOKC * 4);   // residual after attention
  bf16_t* wqkv_t = (bf16_t*)ws_take(p, (size_t)768 * 2304 * 2);  // [2304][768]
  bf16_t* wprj_t = (bf16_t*)ws_take(p, (size_t)768 * 768 * 2);   // [768][768]
  bf16_t* wfc1_t = (bf16_t*)ws_take(p, (size_t)768 * 384 * 2);   // [384][768]
  bf16_t* wfc2_t = (bf16_t*)ws_take(p, (size_t)384 * 768 * 2);   // [768][384]
  bf16_t* ao = h1;
  bf16_t* h2 = qb;
  bf16_t* h3 = kb;

  // 1) weights -> bf16, transposed to [N][K]
  cvt_transpose_bf16<<<(768 * 2304 + 255) / 256, 256, 0, stream>>>(wqkv, wqkv_t, 768, 2304);
  cvt_transpose_bf16<<<(768 * 768 + 255) / 256, 256, 0, stream>>>(wprj, wprj_t, 768, 768);
  cvt_transpose_bf16<<<(768 * 384 + 255) / 256, 256, 0, stream>>>(wfc1, wfc1_t, 768, 384);
  cvt_transpose_bf16<<<(384 * 768 + 255) / 256, 256, 0, stream>>>(wfc2, wfc2_t, 384, 768);

  // 2) LN1
  ln_bf16_kernel<<<M, 256, 0, stream>>>(x, ln1w, ln1b, h1);

  // 3) QKV GEMM -> q/k [B,H,N,D], v [B,H,D,N]
  gemm_bf16_kernel<<<dim3(M / GBM, 2304 / GBN), 256, 0, stream>>>(
      h1, wqkv_t, M, 2304, 768, MODE_QKV, nullptr, nullptr, nullptr, nullptr,
      qb, kb, vb);

  // 4) attention
  attn_kernel<<<8 * 12 * 8, 256, 0, stream>>>(qb, kb, vb, ao);

  // 5) proj GEMM + residual -> x1 fp32
  gemm_bf16_kernel<<<dim3(M / GBM, 768 / GBN), 256, 0, stream>>>(
      ao, wprj_t, M, 768, 768, MODE_PROJ, nullptr, x, x1, nullptr,
      nullptr, nullptr, nullptr);

  // 6) LN2
  ln_bf16_kernel<<<M, 256, 0, stream>>>(x1, ln2w, ln2b, h2);

  // 7) FC1 + bias + exact GELU -> bf16
  gemm_bf16_kernel<<<dim3(M / GBM, 384 / GBN), 256, 0, stream>>>(
      h2, wfc1_t, M, 384, 768, MODE_FC1, bfc1, nullptr, nullptr, h3,
      nullptr, nullptr, nullptr);

  // 8) FC2 + bias + residual -> out fp32
  gemm_bf16_kernel<<<dim3(M / GBM, 768 / GBN), 256, 0, stream>>>(
      h3, wfc2_t, M, 768, 384, MODE_FC2, bfc2, x1, out, nullptr,
      nullptr, nullptr, nullptr);
}
